// SyncMixtralSparseMoeBlock_63007170232537
// MI455X (gfx1250) — matrix operand order repaired and checked by reading, not compiled, not run on hardware
//
#include <hip/hip_runtime.h>
#include <hip/hip_bf16.h>

// ---------------------------------------------------------------------------
// Mixtral sparse MoE block for gfx1250 (MI455X), wave32 + WMMA f16.
//   T=1024 tokens, H=1024, E=8 experts, F=2048, top-2 routing.
//   d_out = [ out (T*H f32) | router_logits (T*8 f32) ]
// ---------------------------------------------------------------------------

#define TT   1024
#define HH   1024
#define EE   8
#define FF   2048
#define FN   128                 // F-chunk processed per outer iteration
#define XSTRIDE (HH + 16)        // f16 elements, keeps 32B alignment, spreads banks
#define ASTRIDE (FN + 16)

typedef __attribute__((ext_vector_type(16))) _Float16 v16h;
typedef __attribute__((ext_vector_type(8)))  float    v8f;

// Load 16 consecutive f32 and convert to a v16h fragment where the 16 halves
// are K-consecutive (B-matrix per-lane layout: 16 contiguous K at lane-group
// offset). 64B aligned source -> 4x b128 loads + packed cvts.
__device__ __forceinline__ v16h load16f_cvt(const float* __restrict__ p) {
    const float4* p4 = (const float4*)p;
    float4 a = p4[0], b = p4[1], c = p4[2], d = p4[3];
    v16h r;
    r[0]=(_Float16)a.x;  r[1]=(_Float16)a.y;  r[2]=(_Float16)a.z;  r[3]=(_Float16)a.w;
    r[4]=(_Float16)b.x;  r[5]=(_Float16)b.y;  r[6]=(_Float16)b.z;  r[7]=(_Float16)b.w;
    r[8]=(_Float16)c.x;  r[9]=(_Float16)c.y;  r[10]=(_Float16)c.z; r[11]=(_Float16)c.w;
    r[12]=(_Float16)d.x; r[13]=(_Float16)d.y; r[14]=(_Float16)d.z; r[15]=(_Float16)d.w;
    return r;
}

// A-matrix fragment (this code's contiguous K order, not the ISA table order):
// lane group g = lane>>4: halves 0..15 -> K = k0 + 16g + i (contiguous, matches B fragment).
__device__ __forceinline__ v16h lds_afrag(const _Float16* __restrict__ row, int k0, int off8) {
    v16h r;
#pragma unroll
    for (int i = 0; i < 8; ++i) r[i] = row[k0 + 2 * off8 + i];
#pragma unroll
    for (int i = 0; i < 8; ++i) r[8 + i] = row[k0 + 2 * off8 + 8 + i];
    return r;
}

// ---------------------------------------------------------------------------
// Router: logits (f32), softmax, top-2, renormalize, build per-expert lists.
// One wave per token. 256 threads = 8 tokens/block.
// ---------------------------------------------------------------------------
__global__ __launch_bounds__(256) void router_kernel(
    const float* __restrict__ x,        // [T,H]
    const float* __restrict__ gate,     // [E,H]
    float* __restrict__ logits_out,     // [T,E]
    int* __restrict__ cnt,              // [E]
    int* __restrict__ tok_idx,          // [E,T]
    float* __restrict__ tok_wgt)        // [E,T]
{
    const int wave = threadIdx.x >> 5;
    const int lane = threadIdx.x & 31;
    const int t = blockIdx.x * 8 + wave;

    const float* xp = x + (size_t)t * HH;
    float xv[32];
#pragma unroll
    for (int j = 0; j < 32; ++j) xv[j] = xp[j * 32 + lane];

    float lg[EE];
#pragma unroll
    for (int e = 0; e < EE; ++e) {
        const float* gp = gate + (size_t)e * HH;
        float s = 0.f;
#pragma unroll
        for (int j = 0; j < 32; ++j) s += xv[j] * gp[j * 32 + lane];
#pragma unroll
        for (int off = 16; off; off >>= 1) s += __shfl_xor(s, off, 32);
        lg[e] = s;   // full reduction: every lane holds the logit
    }

    if (lane == 0) {
#pragma unroll
        for (int e = 0; e < EE; ++e) logits_out[t * EE + e] = lg[e];
        // fp32 softmax over 8
        float m = lg[0];
#pragma unroll
        for (int e = 1; e < EE; ++e) m = fmaxf(m, lg[e]);
        float p[EE];
#pragma unroll
        for (int e = 0; e < EE; ++e) p[e] = __expf(lg[e] - m);
        // top-2, first index wins ties (matches jax.lax.top_k)
        int i1 = 0;
#pragma unroll
        for (int e = 1; e < EE; ++e) if (p[e] > p[i1]) i1 = e;
        int i2 = (i1 == 0) ? 1 : 0;
#pragma unroll
        for (int e = 0; e < EE; ++e) if (e != i1 && e != i2 && p[e] > p[i2]) i2 = e;
        const float denom = p[i1] + p[i2];
        const float wA = p[i1] / denom, wB = p[i2] / denom;

        int pos = atomicAdd(&cnt[i1], 1);
        tok_idx[i1 * TT + pos] = t; tok_wgt[i1 * TT + pos] = wA;
        pos = atomicAdd(&cnt[i2], 1);
        tok_idx[i2 * TT + pos] = t; tok_wgt[i2 * TT + pos] = wB;
    }
}

// ---------------------------------------------------------------------------
// Expert FFN: per (expert, 16-token tile). 8 waves / block.
//   up:   h1 = x W1^T, h3 = x W3^T   (M16 x N16 per wave, K=H via 16x16x32 f16 WMMA)
//   act = silu(h1)*h3*route_w  -> f16 in LDS
//   down: out += act W2^T            (M16 x N1024, K=F chunked by 128)
// ---------------------------------------------------------------------------
__global__ __launch_bounds__(256) void moe_ffn_kernel(
    const float* __restrict__ x,        // [T,H]
    const float* __restrict__ w1,       // [E,F,H]
    const float* __restrict__ w2,       // [E,H,F]
    const float* __restrict__ w3,       // [E,F,H]
    const int*   __restrict__ cnt,      // [E]
    const int*   __restrict__ tok_idx,  // [E,T]
    const float* __restrict__ tok_wgt,  // [E,T]
    float* __restrict__ out)            // [T,H]
{
    const int e    = blockIdx.y;
    const int tile = blockIdx.x;
    const int n_tok = cnt[e];
    if (tile * 16 >= n_tok) return;

    __shared__ _Float16 xlds[16 * XSTRIDE];
    __shared__ _Float16 alds[16 * ASTRIDE];
    __shared__ float wgt_s[16];
    __shared__ int   tok_s[16];

    const int tid  = threadIdx.x;
    const int lane = tid & 31;
    const int wave = tid >> 5;
    const int mrow = lane & 15;         // A row (and B/C column) index
    const int off8  = (lane >> 4) * 8;  // A-frag K sub-offset
    const int off16 = (lane >> 4) * 16; // B-frag K sub-offset
    const int mbase = (lane >> 4) * 8;  // C-layout row base

    if (tid < 16) {
        const int slot = tile * 16 + tid;
        int tk = 0; float wg = 0.f;
        if (slot < n_tok) { tk = tok_idx[e * TT + slot]; wg = tok_wgt[e * TT + slot]; }
        tok_s[tid] = tk; wgt_s[tid] = wg;
    }
    __syncthreads();

    // Stage the 16 token rows as f16 (each wave loads 2 rows).
#pragma unroll
    for (int rr = 0; rr < 2; ++rr) {
        const int r = wave * 2 + rr;
        const float4* src = (const float4*)(x + (size_t)tok_s[r] * HH);
        for (int c = lane; c < HH / 4; c += 32) {
            float4 v = src[c];
            _Float16* d = &xlds[r * XSTRIDE + c * 4];
            d[0] = (_Float16)v.x; d[1] = (_Float16)v.y;
            d[2] = (_Float16)v.z; d[3] = (_Float16)v.w;
        }
    }
    __syncthreads();

    const float* w1e = w1 + (size_t)e * FF * HH;
    const float* w3e = w3 + (size_t)e * FF * HH;
    const float* w2e = w2 + (size_t)e * HH * FF;

    v8f oacc[8];                         // wave covers out cols [wave*128, +128)
#pragma unroll
    for (int j = 0; j < 8; ++j) oacc[j] = (v8f){};

    for (int f0 = 0; f0 < FF; f0 += FN) {
        // ---- up projections: this wave's N-tile = columns [f0+wave*16, +16)
        const int fw = f0 + wave * 16;
        v8f h1 = (v8f){}, h3 = (v8f){};
        const float* bp1 = w1e + (size_t)(fw + mrow) * HH + off16;
        const float* bp3 = w3e + (size_t)(fw + mrow) * HH + off16;
        const _Float16* arow = &xlds[mrow * XSTRIDE];
#pragma unroll 4
        for (int k0 = 0; k0 < HH; k0 += 32) {
            v16h a  = lds_afrag(arow, k0, off8);
            v16h b1 = load16f_cvt(bp1 + k0);
            v16h b3 = load16f_cvt(bp3 + k0);
            h1 = __builtin_amdgcn_wmma_f32_16x16x32_f16(false, a, false, b1,
                                                        (short)0, h1, false, false);
            h3 = __builtin_amdgcn_wmma_f32_16x16x32_f16(false, a, false, b3,
                                                        (short)0, h3, false, false);
        }

        __syncthreads();   // previous chunk's alds readers are done
        // act = silu(h1) * h3 * routing weight, stored f16 into alds
#pragma unroll
        for (int r = 0; r < 8; ++r) {
            const float z = h1[r];
            const float g = z / (1.f + __expf(-z));       // silu
            const float a = g * h3[r] * wgt_s[mbase + r];
            alds[(mbase + r) * ASTRIDE + wave * 16 + mrow] = (_Float16)a;
        }
        __syncthreads();

        // ---- down projection: out[16, wave*128 : +128] += act[16,FN] @ W2^T
        const _Float16* act_row = &alds[mrow * ASTRIDE];
#pragma unroll
        for (int j = 0; j < 8; ++j) {
            const int hw = wave * 128 + j * 16;
            const float* bp2 = w2e + (size_t)(hw + mrow) * FF + f0 + off16;
            v8f acc = oacc[j];
#pragma unroll
            for (int k0 = 0; k0 < FN; k0 += 32) {
                v16h a = lds_afrag(act_row, k0, off8);
                v16h b = load16f_cvt(bp2 + k0);
                acc = __builtin_amdgcn_wmma_f32_16x16x32_f16(false, a, false, b,
                                                             (short)0, acc, false, false);
            }
            oacc[j] = acc;
        }
    }

    // Scatter-accumulate the 16 x 1024 tile. Each token belongs to exactly 2
    // experts; HW f32 atomics, a+b commutative -> deterministic result.
#pragma unroll
    for (int j = 0; j < 8; ++j) {
        const int hcol = wave * 128 + j * 16 + mrow;
#pragma unroll
        for (int r = 0; r < 8; ++r) {
            const int t = tok_s[mbase + r];
            unsafeAtomicAdd(&out[(size_t)t * HH + hcol], oacc[j][r]);
        }
    }
}

// ---------------------------------------------------------------------------
extern "C" void kernel_launch(void* const* d_in, const int* in_sizes, int n_in,
                              void* d_out, int out_size, void* d_ws, size_t ws_size,
                              hipStream_t stream) {
    const float* x    = (const float*)d_in[0];   // hidden_states [1,1024,1024]
    const float* gate = (const float*)d_in[1];   // gate_w [8,1024]
    const float* w1   = (const float*)d_in[2];   // [8,2048,1024]
    const float* w2   = (const float*)d_in[3];   // [8,1024,2048]
    const float* w3   = (const float*)d_in[4];   // [8,2048,1024]

    float* out    = (float*)d_out;               // [T*H]
    float* logits = out + (size_t)TT * HH;       // [T*E]

    // workspace: counts | token indices | token weights  (~64.2 KB)
    int*   cnt     = (int*)d_ws;
    int*   tok_idx = cnt + 32;                   // aligned
    float* tok_wgt = (float*)(tok_idx + EE * TT);

    hipMemsetAsync(cnt, 0, 32 * sizeof(int), stream);
    hipMemsetAsync(out, 0, (size_t)TT * HH * sizeof(float), stream);

    router_kernel<<<TT / 8, 256, 0, stream>>>(x, gate, logits, cnt, tok_idx, tok_wgt);
    moe_ffn_kernel<<<dim3(TT / 16, EE), 256, 0, stream>>>(x, w1, w2, w3,
                                                          cnt, tok_idx, tok_wgt, out);
}
